// Autoencoder__gen204_65481071403328
// MI455X (gfx1250) — matrix operand order repaired and checked by reading, not compiled, not run on hardware
//
#include <hip/hip_runtime.h>

// ---------------------------------------------------------------------------
// MI455X (gfx1250) fused autoencoder + 4-qubit circuit.
// Roofline: mandatory HBM traffic = x (134 MB f32 in) + out (134 MB f32 out)
// => ~11.5 us floor at 23.3 TB/s; 17.8 GFLOP of GEMM is far below the WMMA
// ceiling => memory bound. Streaming tensors (x, out) use non-temporal cache
// policy so the f16 intermediates (~13 MB) and weights stay L2-resident.
// B matrices pre-transposed to [N][K] f16 so WMMA A/B fragments are direct
// contiguous global_load_b128's (CDNA5 ISA 7.12.2 layouts): no LDS, no
// barriers. A sched_barrier(0) fences all fragment loads of a K-step ahead
// of the WMMA chain, so loads clause up in distinct VGPRs and the WMMAs
// retire under incremental s_wait_loadcnt (load t+1.. in flight during
// wmma t) instead of one full-latency wait per WMMA.
// ---------------------------------------------------------------------------

typedef _Float16 v16h __attribute__((ext_vector_type(16)));
typedef _Float16 v8h  __attribute__((ext_vector_type(8)));
typedef float    v8f  __attribute__((ext_vector_type(8)));
typedef float    v4f  __attribute__((ext_vector_type(4)));

union Frag16 { v16h v; v8h h[2]; };

// ---------------------------------------------------------------------------
// Convert f32 weight [K x N] row-major -> f16 transposed [N x K].
// ---------------------------------------------------------------------------
__global__ __launch_bounds__(256) void transpose_f32_to_f16(
    const float* __restrict__ src, _Float16* __restrict__ dst, int K, int N) {
  int idx = blockIdx.x * 256 + threadIdx.x;
  if (idx < K * N) {
    int k = idx / N, n = idx % N;
    dst[(size_t)n * K + k] = (_Float16)src[idx];
  }
}

// ---------------------------------------------------------------------------
// LDS-free WMMA GEMM: C[M x ldn] = act(A[M x K] * Bt[N x K]^T + bias)
//   Block = 256 threads = 8 waves; wave computes 16 rows x NT*16 cols.
//   Fragment layouts per CDNA5 ISA 7.12.2 (16-bit, wave32):
//     A 16x32 : lane r(+hi*16) -> row r halves {K hi*8..+7, K 16+hi*8..+7}
//     B 32x16 : lane r(+hi*16) -> col r halves {K hi*8..+7, K 16+hi*8..+7}
//     C/D f32 : VGPR i -> M = 8*hi + i, N = r
// ---------------------------------------------------------------------------
template <int NT, int K, bool A_F32, bool RELU, bool OUT_F16>
__global__ __launch_bounds__(256) void gemm_wmma(
    const void* __restrict__ Av, const _Float16* __restrict__ Bt,
    const float* __restrict__ bias, void* __restrict__ Cv, int ldn) {
  const int tid  = threadIdx.x;
  const int wave = tid >> 5;
  const int lane = tid & 31;
  const int r    = lane & 15;
  const int hi   = lane >> 4;
  const int m0   = blockIdx.x * 128 + wave * 16;   // wave's 16-row strip
  const int n0   = blockIdx.y * (NT * 16);

  v8f acc[NT];
  v8f zero = {};
#pragma unroll
  for (int t = 0; t < NT; ++t) acc[t] = zero;

  const _Float16* Ah = (const _Float16*)Av + (size_t)(m0 + r) * K;
  const float*    Af = (const float*)Av    + (size_t)(m0 + r) * K;

  for (int k0 = 0; k0 < K; k0 += 32) {
    // ============ load phase: A first (needed earliest), then B[0..NT) =====
    Frag16 a;
    v4f f[4];
    if constexpr (A_F32) {
      // x is streamed once: non-temporal loads keep L2 for f16 intermediates
      const float* p = Af + k0 + hi * 8;
      f[0] = __builtin_nontemporal_load((const v4f*)(p));       // K hi*8..+3
      f[1] = __builtin_nontemporal_load((const v4f*)(p + 4));   // K hi*8+4..+7
      f[2] = __builtin_nontemporal_load((const v4f*)(p + 16));  // K 16+hi*8..+3
      f[3] = __builtin_nontemporal_load((const v4f*)(p + 20));  // K 16+hi*8+4..
    } else {
      a.h[0] = *(const v8h*)(Ah + k0 + hi * 8);
      a.h[1] = *(const v8h*)(Ah + k0 + 16 + hi * 8);
    }
    Frag16 b[NT];
#pragma unroll
    for (int t = 0; t < NT; ++t) {
      const _Float16* bp = Bt + (size_t)(n0 + t * 16 + r) * K + k0 + hi * 8;
      b[t].h[0] = *(const v8h*)(bp);
      b[t].h[1] = *(const v8h*)(bp + 16);
    }
    // Fence: no load may sink below, no WMMA may hoist above. Forces all
    // fragments live in distinct VGPRs -> loads issue as clauses, WMMAs
    // consume under incremental s_wait_loadcnt.
    __builtin_amdgcn_sched_barrier(0);
    // ============ compute phase ===========================================
    if constexpr (A_F32) {
      v8h lo = {(_Float16)f[0].x, (_Float16)f[0].y, (_Float16)f[0].z, (_Float16)f[0].w,
                (_Float16)f[1].x, (_Float16)f[1].y, (_Float16)f[1].z, (_Float16)f[1].w};
      v8h hh = {(_Float16)f[2].x, (_Float16)f[2].y, (_Float16)f[2].z, (_Float16)f[2].w,
                (_Float16)f[3].x, (_Float16)f[3].y, (_Float16)f[3].z, (_Float16)f[3].w};
      a.h[0] = lo;
      a.h[1] = hh;
    }
#pragma unroll
    for (int t = 0; t < NT; ++t)
      acc[t] = __builtin_amdgcn_wmma_f32_16x16x32_f16(
          false, a.v, false, b[t].v, (short)0, acc[t], false, false);
  }

  // ---- epilogue: bias (+ReLU), store f16 or f32(non-temporal) ----
#pragma unroll
  for (int t = 0; t < NT; ++t) {
    const int n  = n0 + t * 16 + r;
    const float bv = bias[n];
#pragma unroll
    for (int i = 0; i < 8; ++i) {
      const int mm = m0 + hi * 8 + i;
      float v = acc[t][i] + bv;
      if (RELU) v = fmaxf(v, 0.0f);
      if (OUT_F16)
        ((_Float16*)Cv)[(size_t)mm * ldn + n] = (_Float16)v;
      else
        __builtin_nontemporal_store(v, &((float*)Cv)[(size_t)mm * ldn + n]);
    }
  }
}

// ---------------------------------------------------------------------------
// Fused middle stage, one row per thread:
//   x4 = h2_row(64) @ enc_w2[:, :4] + enc_b2[:4]   (only 4 latent cols used)
//   4-qubit circuit (RX encode, 3x[Rot x4 + CNOT chain]) -> <Z_q> x4
//   h3 = relu(<Z> @ dec_w0 + dec_b0)   (4 -> 64), stored f16
// ---------------------------------------------------------------------------
__device__ __forceinline__ void apply1(float2* st, int wire, float2 u00,
                                       float2 u01, float2 u10, float2 u11) {
  const int m = 1 << (3 - wire);   // wire 0 is MSB of the 4-bit state index
#pragma unroll
  for (int s = 0; s < 16; ++s) {
    if (s & m) continue;
    float2 a = st[s], b = st[s | m];
    st[s]     = make_float2(u00.x * a.x - u00.y * a.y + u01.x * b.x - u01.y * b.y,
                            u00.x * a.y + u00.y * a.x + u01.x * b.y + u01.y * b.x);
    st[s | m] = make_float2(u10.x * a.x - u10.y * a.y + u11.x * b.x - u11.y * b.y,
                            u10.x * a.y + u10.y * a.x + u11.x * b.y + u11.y * b.x);
  }
}

__global__ __launch_bounds__(256) void quantum_refine(
    const _Float16* __restrict__ h2, const float* __restrict__ w2,
    const float* __restrict__ b2, const float* __restrict__ qw,
    const float* __restrict__ dw0, const float* __restrict__ db0,
    _Float16* __restrict__ h3) {
  const int row = blockIdx.x * 256 + threadIdx.x;
  const _Float16* hr = h2 + (size_t)row * 64;

  float x4[4];
#pragma unroll
  for (int i = 0; i < 4; ++i) {
    float acc = b2[i];
    for (int k = 0; k < 64; ++k) acc += (float)hr[k] * w2[k * 32 + i];
    x4[i] = acc;
  }

  float2 st[16];
#pragma unroll
  for (int s = 0; s < 16; ++s) st[s] = make_float2(0.f, 0.f);
  st[0].x = 1.f;

  // RX(x4[q]) on each wire: [[c, -i s], [-i s, c]]
#pragma unroll
  for (int q = 0; q < 4; ++q) {
    float c = cosf(x4[q] * 0.5f), sn = sinf(x4[q] * 0.5f);
    apply1(st, q, make_float2(c, 0.f), make_float2(0.f, -sn),
                  make_float2(0.f, -sn), make_float2(c, 0.f));
  }

  for (int layer = 0; layer < 3; ++layer) {
#pragma unroll
    for (int q = 0; q < 4; ++q) {
      const float phi = qw[(layer * 4 + q) * 3 + 0];
      const float th  = qw[(layer * 4 + q) * 3 + 1];
      const float om  = qw[(layer * 4 + q) * 3 + 2];
      float hp = 0.5f * (phi + om), hm = 0.5f * (phi - om);
      float ct = cosf(th * 0.5f), stt = sinf(th * 0.5f);
      float chp = cosf(hp), shp = sinf(hp), chm = cosf(hm), shm = sinf(hm);
      // Rot = RZ(om) RY(th) RZ(phi)
      apply1(st, q,
             make_float2(ct * chp, -ct * shp),    // ct * e^{-i(phi+om)/2}
             make_float2(-stt * chm, -stt * shm), // -st * e^{+i(phi-om)/2}
             make_float2(stt * chm, -stt * shm),  //  st * e^{-i(phi-om)/2}
             make_float2(ct * chp, ct * shp));    // ct * e^{+i(phi+om)/2}
    }
    // CNOT chain (q -> q+1): swap amplitudes where control=1
#pragma unroll
    for (int q = 0; q < 3; ++q) {
      const int mc = 1 << (3 - q), mt = 1 << (3 - (q + 1));
#pragma unroll
      for (int s = 0; s < 16; ++s)
        if ((s & mc) && !(s & mt)) {
          float2 tmp = st[s]; st[s] = st[s | mt]; st[s | mt] = tmp;
        }
    }
  }

  float ez[4] = {0.f, 0.f, 0.f, 0.f};
#pragma unroll
  for (int s = 0; s < 16; ++s) {
    float p = st[s].x * st[s].x + st[s].y * st[s].y;
#pragma unroll
    for (int q = 0; q < 4; ++q) ez[q] += ((s >> (3 - q)) & 1) ? -p : p;
  }

  _Float16* out = h3 + (size_t)row * 64;
  for (int j = 0; j < 64; ++j) {
    float a = db0[j];
#pragma unroll
    for (int i = 0; i < 4; ++i) a += ez[i] * dw0[i * 64 + j];
    out[j] = (_Float16)fmaxf(a, 0.f);
  }
}

// ---------------------------------------------------------------------------
// Launch pipeline
// ---------------------------------------------------------------------------
extern "C" void kernel_launch(void* const* d_in, const int* in_sizes, int n_in,
                              void* d_out, int out_size, void* d_ws, size_t ws_size,
                              hipStream_t stream) {
  (void)in_sizes; (void)n_in; (void)out_size; (void)ws_size;
  const float* x      = (const float*)d_in[0];
  const float* enc_w0 = (const float*)d_in[1];
  const float* enc_b0 = (const float*)d_in[2];
  const float* enc_w1 = (const float*)d_in[3];
  const float* enc_b1 = (const float*)d_in[4];
  const float* enc_w2 = (const float*)d_in[5];
  const float* enc_b2 = (const float*)d_in[6];
  const float* qw     = (const float*)d_in[7];
  const float* dec_w0 = (const float*)d_in[8];
  const float* dec_b0 = (const float*)d_in[9];
  const float* dec_w1 = (const float*)d_in[10];
  const float* dec_b1 = (const float*)d_in[11];
  const float* dec_w2 = (const float*)d_in[12];
  const float* dec_b2 = (const float*)d_in[13];

  // workspace layout (256B-aligned offsets), total ~13.1 MB (L2-resident)
  char* ws = (char*)d_ws;
  _Float16* bt0  = (_Float16*)(ws + 0);         // enc_w0^T  [128 x 2048] 512 KB
  _Float16* bt1  = (_Float16*)(ws + 524288);    // enc_w1^T  [64 x 128]    16 KB
  _Float16* btd1 = (_Float16*)(ws + 540672);    // dec_w1^T  [128 x 64]    16 KB
  _Float16* btd2 = (_Float16*)(ws + 557056);    // dec_w2^T  [2048 x 128] 512 KB
  _Float16* h1h  = (_Float16*)(ws + 1081344);   // [16384 x 128]            4 MB
  _Float16* h2h  = (_Float16*)(ws + 5275648);   // [16384 x 64]             2 MB
  _Float16* h3h  = (_Float16*)(ws + 7372800);   // [16384 x 64]             2 MB
  _Float16* h4h  = (_Float16*)(ws + 9469952);   // [16384 x 128]            4 MB

  // weight converts: f32 [K x N] -> f16 transposed [N x K]
  transpose_f32_to_f16<<<dim3((2048 * 128) / 256), 256, 0, stream>>>(enc_w0, bt0, 2048, 128);
  transpose_f32_to_f16<<<dim3((128 * 64) / 256),   256, 0, stream>>>(enc_w1, bt1, 128, 64);
  transpose_f32_to_f16<<<dim3((64 * 128) / 256),   256, 0, stream>>>(dec_w1, btd1, 64, 128);
  transpose_f32_to_f16<<<dim3((128 * 2048) / 256), 256, 0, stream>>>(dec_w2, btd2, 128, 2048);

  // encoder: h1 = relu(x @ W0 + b0)      [16384 x 2048] -> [16384 x 128]
  gemm_wmma<8, 2048, true, true, true>
      <<<dim3(128, 1), 256, 0, stream>>>(x, bt0, enc_b0, h1h, 128);
  // encoder: h2 = relu(h1 @ W1 + b1)     [16384 x 128] -> [16384 x 64]
  gemm_wmma<4, 128, false, true, true>
      <<<dim3(128, 1), 256, 0, stream>>>(h1h, bt1, enc_b1, h2h, 64);
  // latent(4 cols) -> circuit -> dec layer0   [16384 x 64] -> [16384 x 64]
  quantum_refine<<<dim3(16384 / 256), 256, 0, stream>>>(h2h, enc_w2, enc_b2, qw,
                                                        dec_w0, dec_b0, h3h);
  // decoder: h4 = relu(h3 @ dW1 + db1)   [16384 x 64] -> [16384 x 128]
  gemm_wmma<8, 64, false, true, true>
      <<<dim3(128, 1), 256, 0, stream>>>(h3h, btd1, dec_b1, h4h, 128);
  // decoder: out = h4 @ dW2 + db2        [16384 x 128] -> [16384 x 2048] f32
  gemm_wmma<8, 128, false, false, false>
      <<<dim3(128, 16), 256, 0, stream>>>(h4h, btd2, dec_b2, d_out, 2048);
}
